// NonlocalWeightedAverage_4243427689211
// MI455X (gfx1250) — matrix operand order repaired and checked by reading, not compiled, not run on hardware
//
#include <hip/hip_runtime.h>
#include <hip/hip_bf16.h>

// Shapes fixed by the reference: B=4, C=64, H=W=64, PATCH=3, ALPHA=0.1
#define HH 64
#define WWID 64
#define HW 4096
#define CC 64
#define DD 576            // C * 3 * 3
#define KSTEPS 18         // DD / 32 (bf16 WMMA K=32)
#define BATCH 4
// softmax computed in base-2 domain: s2 = corr * (1/alpha) * log2(e)
#define SCALE2 14.426950408889634f

typedef __attribute__((ext_vector_type(16))) __bf16 v16bf;
typedef __attribute__((ext_vector_type(8)))  float  v8f;
typedef __attribute__((ext_vector_type(4)))  unsigned int v4u;

struct Pair128 { v4u lo, hi; };

static __device__ __forceinline__ v16bf make_frag(v4u lo, v4u hi) {
    Pair128 p{lo, hi};
    return __builtin_bit_cast(v16bf, p);
}

// ---------------------------------------------------------------------------
// Kernel 1: build bf16 unfolded descriptor matrix kmat[b][p][d], d = patch*64+c
// (zero-padded 3x3 unfold of feature [B,64,64,64]; descriptor ordering is
//  permutation-invariant under the corr contraction.)
// ---------------------------------------------------------------------------
__global__ void build_kmat(const float* __restrict__ feature,
                           __bf16* __restrict__ kmat) {
    int tid = blockIdx.x * blockDim.x + threadIdx.x;   // exact grid, no tail
    int d   = tid % DD;
    int rem = tid / DD;
    int p   = rem % HW;
    int b   = rem / HW;
    int patch = d / CC;
    int c     = d % CC;
    int h = p >> 6, w = p & 63;
    int y = h + patch / 3 - 1;
    int x = w + patch % 3 - 1;
    float v = 0.0f;
    if ((unsigned)y < HH && (unsigned)x < WWID)
        v = feature[(((size_t)b * CC + c) * HH + y) * WWID + x];
    kmat[tid] = (__bf16)v;
}

// ---------------------------------------------------------------------------
// Kernel 2: fused flash-attention over the self-correlation.
//   grid = 128 blocks (4 batches x 32 row-groups), 256 threads = 8 waves.
//   wave -> 16 query rows; A fragments register-resident; TWO column tiles
//   per iteration (independent WMMA accumulator chains c0/c1 interleaved);
//   online softmax in base-2 domain with butterfly max shared across the
//   32-column pair; rgb accumulation fused.
// ---------------------------------------------------------------------------
__global__ void __launch_bounds__(256, 1)
nonlocal_attn(const float* __restrict__ x_rgb,
              const __bf16* __restrict__ kmat,
              float* __restrict__ out) {
    const int lane    = threadIdx.x & 31;
    const int wave    = threadIdx.x >> 5;
    const int b       = blockIdx.x >> 5;              // 32 row-groups per batch
    const int rowTile = (blockIdx.x & 31) * 128 + wave * 16;
    const int laneLow = lane & 15;
    const int hiHalf  = lane >> 4;                    // 0 or 1

    const __bf16* kb = kmat + (size_t)b * HW * DD;

    // --- A fragments: 16x32 bf16 per k-step. contiguous k order used here (same as B):
    //     lane<16: halves 0-15 -> K+0..15
    //     lane>=16: halves 0-15 -> K+16..31
    // Row per lane = lane&15; each half-group is one contiguous 16B load.
    const v4u* arow = (const v4u*)(kb + (size_t)(rowTile + laneLow) * DD); // 72 x 16B/row
    const int sA = hiHalf * 2;   // +0/+2 (16B units) inside each 64B k-group
    const int sB = hiHalf * 2;   // B: lanes>=16 start at K+16 -> +32B

    v16bf afrag[KSTEPS];
#pragma unroll
    for (int k = 0; k < KSTEPS; ++k) {
        v4u lo = arow[k * 4 + sA];
        v4u hi = arow[k * 4 + sA + 1];
        afrag[k] = make_frag(lo, hi);
    }

    const float* rgbb = x_rgb + (size_t)b * 3 * HW;

    // Online-softmax state (base-2 domain). C layout: VGPR r -> row (r+8*hiHalf),
    // lane -> column (lane&15). lsum/acc are per-lane partials over columns.
    float m[8], lsum[8], aR[8], aG[8], aB[8];
#pragma unroll
    for (int r = 0; r < 8; ++r) {
        m[r] = -INFINITY; lsum[r] = 0.f; aR[r] = 0.f; aG[r] = 0.f; aB[r] = 0.f;
    }

    for (int pair = 0; pair < HW / 32; ++pair) {
        const int q0 = pair * 32 + laneLow;
        const int q1 = q0 + 16;
        // --- B fragments: 32x16 bf16. Lane = column; K contiguous per lane:
        //     lanes<16 -> K+0..15, lanes>=16 -> K+16..31 (32 contiguous bytes).
        const v4u* brow0 = (const v4u*)(kb + (size_t)q0 * DD);
        const v4u* brow1 = (const v4u*)(kb + (size_t)q1 * DD);

        v8f c0 = {};
        v8f c1 = {};
#pragma unroll
        for (int k = 0; k < KSTEPS; ++k) {
            v4u l0 = brow0[k * 4 + sB];
            v4u h0 = brow0[k * 4 + sB + 1];
            v4u l1 = brow1[k * 4 + sB];
            v4u h1 = brow1[k * 4 + sB + 1];
            v16bf bf0 = make_frag(l0, h0);
            v16bf bf1 = make_frag(l1, h1);
            c0 = __builtin_amdgcn_wmma_f32_16x16x32_bf16(
                     false, afrag[k], false, bf0, (short)0, c0, false, false);
            c1 = __builtin_amdgcn_wmma_f32_16x16x32_bf16(
                     false, afrag[k], false, bf1, (short)0, c1, false, false);
        }

        const float rq0 = rgbb[q0];
        const float gq0 = rgbb[HW + q0];
        const float bq0 = rgbb[2 * HW + q0];
        const float rq1 = rgbb[q1];
        const float gq1 = rgbb[HW + q1];
        const float bq1 = rgbb[2 * HW + q1];

#pragma unroll
        for (int r = 0; r < 8; ++r) {
            float s0 = c0[r] * SCALE2;
            float s1 = c1[r] * SCALE2;
            // per-row max across the 32 columns of the pair: per-lane max
            // first, then one 16-lane butterfly (masks<16 stay inside the
            // lane-half that owns this row set)
            float t = fmaxf(s0, s1);
#pragma unroll
            for (int mk = 1; mk < 16; mk <<= 1)
                t = fmaxf(t, __shfl_xor(t, mk, 32));
            float nm = fmaxf(m[r], t);
            float f  = exp2f(m[r] - nm);    // 0 on first pair (m = -inf)
            float p0 = exp2f(s0 - nm);
            float p1 = exp2f(s1 - nm);
            lsum[r] = lsum[r] * f + (p0 + p1);
            aR[r]   = aR[r]   * f + (p0 * rq0 + p1 * rq1);
            aG[r]   = aG[r]   * f + (p0 * gq0 + p1 * gq1);
            aB[r]   = aB[r]   * f + (p0 * bq0 + p1 * bq1);
            m[r] = nm;
        }
    }

    // Final reduction of per-lane partials across the 16-lane column groups.
#pragma unroll
    for (int r = 0; r < 8; ++r) {
#pragma unroll
        for (int mk = 1; mk < 16; mk <<= 1) {
            lsum[r] += __shfl_xor(lsum[r], mk, 32);
            aR[r]   += __shfl_xor(aR[r],   mk, 32);
            aG[r]   += __shfl_xor(aG[r],   mk, 32);
            aB[r]   += __shfl_xor(aB[r],   mk, 32);
        }
    }

    if (laneLow == 0) {
        float* ob = out + (size_t)b * 3 * HW;
#pragma unroll
        for (int r = 0; r < 8; ++r) {
            int row = rowTile + r + 8 * hiHalf;
            float inv = 1.0f / lsum[r];
            ob[row]          = aR[r] * inv;
            ob[HW + row]     = aG[r] * inv;
            ob[2 * HW + row] = aB[r] * inv;
        }
    }
}

// ---------------------------------------------------------------------------
extern "C" void kernel_launch(void* const* d_in, const int* in_sizes, int n_in,
                              void* d_out, int out_size, void* d_ws, size_t ws_size,
                              hipStream_t stream) {
    const float* x_rgb   = (const float*)d_in[0];   // [4,3,64,64]
    const float* feature = (const float*)d_in[1];   // [4,64,64,64]
    __bf16* kmat = (__bf16*)d_ws;                   // needs 4*4096*576*2 = 18.9 MB

    const int total = BATCH * HW * DD;              // 9,437,184 (divisible by 256)
    build_kmat<<<total / 256, 256, 0, stream>>>(feature, kmat);

    nonlocal_attn<<<dim3(BATCH * 32), dim3(256), 0, stream>>>(
        x_rgb, kmat, (float*)d_out);
}
